// Attention_25769804179
// MI455X (gfx1250) — matrix operand order repaired and checked by reading, not compiled, not run on hardware
//
#include <hip/hip_runtime.h>

typedef _Float16 half_t;
typedef _Float16 v16h __attribute__((ext_vector_type(16)));
typedef _Float16 v8h  __attribute__((ext_vector_type(8)));
typedef _Float16 v4h  __attribute__((ext_vector_type(4)));
typedef float    v8f  __attribute__((ext_vector_type(8)));
typedef int      i32x4 __attribute__((ext_vector_type(4)));

#define DIM    1024
#define NHEADS 16
#define HDIM   64
#define SEQ    2048
#define BATCH  4
#define MROWS  (BATCH * SEQ)   /* 8192 */
#define QK_SCALE 0.125f        /* 64^-0.5 */

#if __has_builtin(__builtin_amdgcn_global_load_async_to_lds_b128) && \
    __has_builtin(__builtin_amdgcn_s_wait_asynccnt)
#define HAVE_ASYNC_LDS 1
#else
#define HAVE_ASYNC_LDS 0
#endif

/* ---------------- WMMA helpers (CDNA5, wave32) ---------------- */

static __device__ __forceinline__ v8f wmma16x16x32(v16h a, v16h b, v8f c) {
  /* D = A(16x32 f16) * B(32x16 f16) + C(16x16 f32) */
  return __builtin_amdgcn_wmma_f32_16x16x32_f16(false, a, false, b,
                                                (short)0, c, false, false);
}

/* A fragment, 16x32 f16, row-major source with leading dim `ld` (halves).
   Contiguous k order used here: lanes 0-15 -> M=lane, K=0..15; lanes 16-31 -> K=16..31. */
static __device__ __forceinline__ v16h load_frag_a(const half_t* p, int ld) {
  const int lane = threadIdx.x & 31;
  const int m  = lane & 15;
  const int kb = (lane & 16) ? 16 : 0;
  const half_t* q = p + (size_t)m * ld + kb;
  v8h lo = *(const v8h*)(q);        /* K = kb .. kb+7   */
  v8h hi = *(const v8h*)(q + 8);    /* K = kb+8..kb+15  */
  return __builtin_shufflevector(lo, hi, 0,1,2,3,4,5,6,7,8,9,10,11,12,13,14,15);
}

/* B fragment, 32x16 f16, from a TRANSPOSED source BT[n][k] with leading dim `ld`.
   Contiguous k order used here: lanes 0-15 -> N=lane, K=0..15; lanes 16-31 -> K=16..31. */
static __device__ __forceinline__ v16h load_frag_bT(const half_t* p, int ld) {
  const int lane = threadIdx.x & 31;
  const int n  = lane & 15;
  const int kb = (lane & 16) ? 16 : 0;
  const half_t* q = p + (size_t)n * ld + kb;
  v8h lo = *(const v8h*)(q);
  v8h hi = *(const v8h*)(q + 8);
  return __builtin_shufflevector(lo, hi, 0,1,2,3,4,5,6,7,8,9,10,11,12,13,14,15);
}

static __device__ __forceinline__ void wait_lds() {
#if __has_builtin(__builtin_amdgcn_s_wait_dscnt)
  __builtin_amdgcn_s_wait_dscnt(0);
#endif
}

/* one 16-byte global -> LDS copy; async (ASYNCcnt) when available */
static __device__ __forceinline__ void copy16_g2lds(half_t* lds_dst, const half_t* gsrc) {
#if HAVE_ASYNC_LDS
  __builtin_amdgcn_global_load_async_to_lds_b128((i32x4*)gsrc, (i32x4*)lds_dst, 0, 0);
#else
  *(v8h*)lds_dst = *(const v8h*)gsrc;
#endif
}

static __device__ __forceinline__ void wait_async() {
#if HAVE_ASYNC_LDS
  __builtin_amdgcn_s_wait_asynccnt(0);
#endif
}

/* ---------------- elementwise prep kernels ---------------- */

__global__ void k_cvt4(const float* __restrict__ in, half_t* __restrict__ out, int n4) {
  int i = blockIdx.x * blockDim.x + threadIdx.x;
  if (i < n4) {
    float4 v = ((const float4*)in)[i];
    v4h o;
    o[0] = (half_t)v.x; o[1] = (half_t)v.y; o[2] = (half_t)v.z; o[3] = (half_t)v.w;
    ((v4h*)out)[i] = o;
  }
}

/* in[rows][cols] f32  ->  out[cols][rows] f16 */
__global__ void k_transpose_cvt(const float* __restrict__ in, half_t* __restrict__ out,
                                int rows, int cols) {
  int idx = blockIdx.x * blockDim.x + threadIdx.x;
  if (idx < rows * cols) {
    int r = idx / cols, c = idx - r * cols;
    out[(size_t)c * rows + r] = (half_t)in[idx];
  }
}

/* ---------------- QKV GEMM: [8192x1024] x [1024x3072] + bias -> Q/K/V head-major f16 --- */

__global__ __launch_bounds__(256) void k_gemm_qkv(
    const half_t* __restrict__ A,   /* xh  [M][K]  */
    const half_t* __restrict__ BT,  /* WqkvT [N][K] */
    const float*  __restrict__ bias,
    half_t* __restrict__ Qh, half_t* __restrict__ Kh, half_t* __restrict__ Vh) {
  const int wave = threadIdx.x >> 5, lane = threadIdx.x & 31;
  const int wm = wave >> 1, wn = wave & 1;
  const int mbase = blockIdx.x * 128 + wm * 32;
  const int nbase = blockIdx.y * 128 + wn * 64;

  v8f acc[2][4] = {};
  for (int k0 = 0; k0 < DIM; k0 += 32) {
    v16h a0 = load_frag_a(A + (size_t)mbase * DIM + k0, DIM);
    v16h a1 = load_frag_a(A + (size_t)(mbase + 16) * DIM + k0, DIM);
#pragma unroll
    for (int j = 0; j < 4; j++) {
      v16h bf = load_frag_bT(BT + (size_t)(nbase + 16 * j) * DIM + k0, DIM);
      acc[0][j] = wmma16x16x32(a0, bf, acc[0][j]);
      acc[1][j] = wmma16x16x32(a1, bf, acc[1][j]);
    }
  }

  const int ncol = lane & 15, rbase = (lane >> 4) * 8;
#pragma unroll
  for (int i = 0; i < 2; i++)
#pragma unroll
    for (int j = 0; j < 4; j++) {
      int n   = nbase + 16 * j + ncol;
      int sel = n >> 10;            /* 0:Q 1:K 2:V   (uniform per 16-wide tile) */
      int h   = (n >> 6) & 15;
      int d   = n & 63;
      float bb = bias[n];
      half_t* dst = (sel == 0) ? Qh : ((sel == 1) ? Kh : Vh);
      float sc = (sel == 0) ? QK_SCALE : 1.0f;
#pragma unroll
      for (int r = 0; r < 8; r++) {
        int row  = mbase + 16 * i + rbase + r;
        int bidx = row >> 11, spos = row & 2047;
        size_t o = (((size_t)(bidx * NHEADS + h)) * SEQ + spos) * HDIM + d;
        dst[o] = (half_t)((acc[i][j][r] + bb) * sc);
      }
    }
}

/* ---------------- Flash attention ----------------
   grid = (S/128, B*H); 256 threads = 8 waves, 16 q-rows each.
   V tiles (64 keys) stream through a double-buffered raw LDS stage filled with
   async global->LDS copies, then get transposed LDS->LDS for B-fragment reads. */

__global__ __launch_bounds__(256) void k_flash(
    const half_t* __restrict__ Qh, const half_t* __restrict__ Kh,
    const half_t* __restrict__ Vh, half_t* __restrict__ AO /* [8192][1024] f16 */) {
  __shared__ half_t sVraw[2][64 * 64];   /* raw V tiles [key][hd], double buffered */
  __shared__ half_t sVt[64 * 72];        /* V tile transposed: [hd][key], padded   */
  __shared__ half_t sP[8 * 16 * 72];     /* per-wave P scratch: [16 rows][64 keys] */

  const int tid  = threadIdx.x;
  const int w    = tid >> 5;
  const int lane = tid & 31;
  const int bh   = blockIdx.y;                  /* 0..63  (b*16 + h) */
  const int qrow0 = blockIdx.x * 128 + w * 16;  /* q rows within (b,h) */

  const half_t* Qb = Qh + (size_t)bh * SEQ * HDIM;
  const half_t* Kb = Kh + (size_t)bh * SEQ * HDIM;
  const half_t* Vb = Vh + (size_t)bh * SEQ * HDIM;

  /* Q is pre-scaled by hd^-0.5 */
  v16h aq0 = load_frag_a(Qb + (size_t)qrow0 * HDIM,      HDIM);
  v16h aq1 = load_frag_a(Qb + (size_t)qrow0 * HDIM + 32, HDIM);

  v8f acc[4] = {};
  float m_i[8], l_i[8];
#pragma unroll
  for (int r = 0; r < 8; r++) { m_i[r] = -1e30f; l_i[r] = 0.f; }

  half_t* Pw = sP + w * (16 * 72);
  const int ncol = lane & 15;
  const int rbase = (lane >> 4) * 8;

  /* prologue: kick off async fill of V tile 0 (512 x 16B chunks, 2 per thread) */
  for (int c = tid; c < 512; c += 256)
    copy16_g2lds(&sVraw[0][c * 8], Vb + (size_t)c * 8);

  for (int kt = 0; kt < SEQ; kt += 64) {
    const int buf = (kt >> 6) & 1;
    wait_async();        /* this wave's async fills of sVraw[buf] done */
    __syncthreads();     /* all waves' fills done                       */

    /* start async fill of the NEXT tile while we compute on this one */
    if (kt + 64 < SEQ)
      for (int c = tid; c < 512; c += 256)
        copy16_g2lds(&sVraw[buf ^ 1][c * 8], Vb + (size_t)(kt + 64) * HDIM + c * 8);

    /* transpose raw tile -> sVt[d][key] (LDS->LDS through registers) */
    for (int e = tid; e < 512; e += 256) {
      int key = e >> 3;
      int d0  = (e & 7) * 8;
      v8h v = *(const v8h*)(&sVraw[buf][key * HDIM + d0]);
#pragma unroll
      for (int t = 0; t < 8; t++) sVt[(d0 + t) * 72 + key] = v[t];
    }
    if (kt + 64 < SEQ)   /* warm L2/WGP$ for the next K tile */
      __builtin_prefetch(Kb + (size_t)(kt + 64) * HDIM + tid * 32, 0, 1);
    __syncthreads();

    /* scores S = Q * K^T  (K stored [key][hd] == transposed-B layout) */
    v8f s[4];
#pragma unroll
    for (int j = 0; j < 4; j++) {
      v8f z = {};
      v16h bk0 = load_frag_bT(Kb + (size_t)(kt + 16 * j) * HDIM,      HDIM);
      v16h bk1 = load_frag_bT(Kb + (size_t)(kt + 16 * j) * HDIM + 32, HDIM);
      z = wmma16x16x32(aq0, bk0, z);
      z = wmma16x16x32(aq1, bk1, z);
      s[j] = z;
    }

    /* online softmax: rows live as (VGPR r, lane half) in the C layout */
#pragma unroll
    for (int r = 0; r < 8; r++) {
      float t = fmaxf(fmaxf(s[0][r], s[1][r]), fmaxf(s[2][r], s[3][r]));
#pragma unroll
      for (int off = 1; off < 16; off <<= 1) t = fmaxf(t, __shfl_xor(t, off, 32));
      float mn   = fmaxf(m_i[r], t);
      float corr = __expf(m_i[r] - mn);
      m_i[r] = mn;
      l_i[r] *= corr;
#pragma unroll
      for (int j = 0; j < 4; j++) acc[j][r] = acc[j][r] * corr;
      float rs = 0.f;
#pragma unroll
      for (int j = 0; j < 4; j++) {
        float p = __expf(s[j][r] - mn);
        s[j][r] = p;
        rs += p;
      }
#pragma unroll
      for (int off = 1; off < 16; off <<= 1) rs += __shfl_xor(rs, off, 32);
      l_i[r] += rs;
    }

    /* C-layout -> A-layout transpose of P through per-wave LDS (DS in-order per wave) */
#pragma unroll
    for (int j = 0; j < 4; j++)
#pragma unroll
      for (int r = 0; r < 8; r++)
        Pw[(rbase + r) * 72 + j * 16 + ncol] = (half_t)s[j][r];
    wait_lds();

    v16h ap0 = load_frag_a(Pw,      72);
    v16h ap1 = load_frag_a(Pw + 32, 72);
#pragma unroll
    for (int j2 = 0; j2 < 4; j2++) {
      v16h bv0 = load_frag_bT(sVt + (16 * j2) * 72,      72);
      v16h bv1 = load_frag_bT(sVt + (16 * j2) * 72 + 32, 72);
      acc[j2] = wmma16x16x32(ap0, bv0, acc[j2]);
      acc[j2] = wmma16x16x32(ap1, bv1, acc[j2]);
    }
    __syncthreads();   /* all waves done with sVt before it is restaged */
  }

  /* normalize and scatter to [B*S][D] f16 (column = h*64 + d) */
  const int b = bh >> 4, h = bh & 15;
#pragma unroll
  for (int r = 0; r < 8; r++) {
    float inv = 1.f / l_i[r];
    int row = b * SEQ + qrow0 + rbase + r;
#pragma unroll
    for (int j = 0; j < 4; j++)
      AO[(size_t)row * DIM + h * HDIM + j * 16 + ncol] = (half_t)(acc[j][r] * inv);
  }
}

/* ---------------- output projection: [8192x1024] x [1024x1024] + bias -> f32 -------- */

__global__ __launch_bounds__(256) void k_gemm_proj(
    const half_t* __restrict__ A,   /* attn-out [M][K] f16 */
    const half_t* __restrict__ BT,  /* WprojT [N][K] f16   */
    const float*  __restrict__ bias,
    float* __restrict__ out) {
  const int wave = threadIdx.x >> 5, lane = threadIdx.x & 31;
  const int wm = wave >> 1, wn = wave & 1;
  const int mbase = blockIdx.x * 128 + wm * 32;
  const int nbase = blockIdx.y * 128 + wn * 64;

  v8f acc[2][4] = {};
  for (int k0 = 0; k0 < DIM; k0 += 32) {
    v16h a0 = load_frag_a(A + (size_t)mbase * DIM + k0, DIM);
    v16h a1 = load_frag_a(A + (size_t)(mbase + 16) * DIM + k0, DIM);
#pragma unroll
    for (int j = 0; j < 4; j++) {
      v16h bf = load_frag_bT(BT + (size_t)(nbase + 16 * j) * DIM + k0, DIM);
      acc[0][j] = wmma16x16x32(a0, bf, acc[0][j]);
      acc[1][j] = wmma16x16x32(a1, bf, acc[1][j]);
    }
  }

  const int ncol = lane & 15, rbase = (lane >> 4) * 8;
#pragma unroll
  for (int i = 0; i < 2; i++)
#pragma unroll
    for (int j = 0; j < 4; j++) {
      int n = nbase + 16 * j + ncol;
      float bb = bias[n];
#pragma unroll
      for (int r = 0; r < 8; r++) {
        int row = mbase + 16 * i + rbase + r;
        out[(size_t)row * DIM + n] = acc[i][j][r] + bb;
      }
    }
}

/* ---------------- launcher ---------------- */

extern "C" void kernel_launch(void* const* d_in, const int* in_sizes, int n_in,
                              void* d_out, int out_size, void* d_ws, size_t ws_size,
                              hipStream_t stream) {
  (void)in_sizes; (void)n_in; (void)out_size; (void)ws_size;

  const float* x     = (const float*)d_in[0];
  const float* Wqkv  = (const float*)d_in[1];
  const float* bqkv  = (const float*)d_in[2];
  const float* Wproj = (const float*)d_in[3];
  const float* bproj = (const float*)d_in[4];
  float* out = (float*)d_out;

  /* f16 workspace carve-up (all offsets 256B-aligned), total 92,274,688 B */
  char* ws = (char*)d_ws;
  half_t* xh     = (half_t*)(ws + 0);          /* 8192*1024        = 16 MiB */
  half_t* wqkvT  = (half_t*)(ws + 16777216);   /* 3072*1024 f16    =  6 MiB */
  half_t* wprojT = (half_t*)(ws + 23068672);   /* 1024*1024 f16    =  2 MiB */
  half_t* Qh     = (half_t*)(ws + 25165824);   /* head-major, pre-scaled    */
  half_t* Kh     = (half_t*)(ws + 41943040);
  half_t* Vh     = (half_t*)(ws + 58720256);
  half_t* AO     = (half_t*)(ws + 75497472);   /* attention out [8192][1024] */

  /* 1) precision/layout prep */
  k_cvt4<<<8192, 256, 0, stream>>>(x, xh, (MROWS * DIM) / 4);
  k_transpose_cvt<<<(DIM * 3 * DIM + 255) / 256, 256, 0, stream>>>(Wqkv, wqkvT, DIM, 3 * DIM);
  k_transpose_cvt<<<(DIM * DIM + 255) / 256, 256, 0, stream>>>(Wproj, wprojT, DIM, DIM);

  /* 2) QKV projection + head split (+ Q pre-scale) */
  k_gemm_qkv<<<dim3(MROWS / 128, (3 * DIM) / 128), 256, 0, stream>>>(
      xh, wqkvT, bqkv, Qh, Kh, Vh);

  /* 3) flash attention: grid = (S/128 q-tiles, B*H) */
  k_flash<<<dim3(SEQ / 128, BATCH * NHEADS), 256, 0, stream>>>(Qh, Kh, Vh, AO);

  /* 4) output projection to f32 */
  k_gemm_proj<<<dim3(MROWS / 128, DIM / 128), 256, 0, stream>>>(AO, wprojT, bproj, out);
}
